// RWRLayer_28080496181629
// MI455X (gfx1250) — compile-verified, matrix operand order checked by reading, not run on hardware
//
#include <hip/hip_runtime.h>
#include <hip/hip_bf16.h>

typedef __attribute__((ext_vector_type(16))) _Float16 v16h;
typedef __attribute__((ext_vector_type(8)))  _Float16 v8h;
typedef __attribute__((ext_vector_type(8)))  float    v8f;
typedef __attribute__((ext_vector_type(4)))  float    v4f;
typedef __attribute__((ext_vector_type(4)))  int      v4i;

#define N_NODES 8192
#define FIN     512
#define FOUT    64
#define GAT_ALPHA 0.2f
#define GAT_NEG_INF -9.0e15f

__device__ __forceinline__ float lrelu(float x) { return x >= 0.f ? x : GAT_ALPHA * x; }

// ---------------- kernel 0: W (512x64 f32, row-major) -> Wt (64x512 f16) ----------------
__global__ void k_wt(const float* __restrict__ W, _Float16* __restrict__ Wt) {
    int idx = blockIdx.x * 256 + threadIdx.x;   // 0..32767
    int k = idx >> 6, col = idx & 63;
    Wt[col * FIN + k] = (_Float16)W[idx];
}

// ---------------- kernel 1: h = x@W via WMMA; emit h_t (f16, 64x8192) and t1,t2 ----------
// block = 16 rows, 128 threads (4 waves). wave w owns output col-tile n=w (16 cols).
__global__ void __launch_bounds__(128) k_h(const float* __restrict__ x,
                                           const _Float16* __restrict__ Wt,
                                           const float* __restrict__ a_vec,
                                           float* __restrict__ t1,
                                           float* __restrict__ t2,
                                           _Float16* __restrict__ ht) {
    __shared__ float s_t1[16], s_t2[16];
    const int tid  = threadIdx.x;
    const int lane = tid & 31, wv = tid >> 5;
    const int r = lane & 15, g = lane >> 4;
    const int row0 = blockIdx.x * 16;

    if (tid < 16) { s_t1[tid] = 0.f; s_t2[tid] = 0.f; }
    __syncthreads();

    const float*    xrow  = x  + (size_t)(row0 + r) * FIN;
    const _Float16* wtrow = Wt + (size_t)(16 * wv + r) * FIN;

    v8f acc{};
    for (int k = 0; k < FIN; k += 32) {
        const int ja = k + 8 * g;          // K run 0: [ja, ja+8)
        const int jb = k + 16 + 8 * g;     // K run 1: [jb, jb+8)
        // x is streamed once -> non-temporal, keep L2 for reused data
        v4f x0 = __builtin_nontemporal_load((const v4f*)(xrow + ja));
        v4f x1 = __builtin_nontemporal_load((const v4f*)(xrow + ja + 4));
        v4f x2 = __builtin_nontemporal_load((const v4f*)(xrow + jb));
        v4f x3 = __builtin_nontemporal_load((const v4f*)(xrow + jb + 4));
        v16h a;
#pragma unroll
        for (int i = 0; i < 4; ++i) {
            a[i]      = (_Float16)x0[i];
            a[4 + i]  = (_Float16)x1[i];
            a[8 + i]  = (_Float16)x2[i];
            a[12 + i] = (_Float16)x3[i];
        }
        v8h blo = *(const v8h*)(wtrow + ja);
        v8h bhi = *(const v8h*)(wtrow + jb);
        v16h b = __builtin_shufflevector(blo, bhi, 0,1,2,3,4,5,6,7,8,9,10,11,12,13,14,15);
        acc = __builtin_amdgcn_wmma_f32_16x16x32_f16(false, a, false, b, (short)0, acc, false, false);
    }

    // acc[v] = h[row0 + v + 8g, 16*wv + r]; store transposed h_t[col][row] (rows contiguous)
    v8h hv;
#pragma unroll
    for (int v = 0; v < 8; ++v) hv[v] = (_Float16)acc[v];
    *(v8h*)(ht + (size_t)(16 * wv + r) * N_NODES + row0 + 8 * g) = hv;

    const float a1c = a_vec[16 * wv + r];
    const float a2c = a_vec[FOUT + 16 * wv + r];
#pragma unroll
    for (int v = 0; v < 8; ++v) {
        atomicAdd(&s_t1[v + 8 * g], acc[v] * a1c);
        atomicAdd(&s_t2[v + 8 * g], acc[v] * a2c);
    }
    __syncthreads();
    if (tid < 16)       t1[row0 + tid]      = s_t1[tid];
    else if (tid < 32)  t2[row0 + tid - 16] = s_t2[tid - 16];
}

// ---------------- kernel 2: T2max = max(t2) ----------------
__global__ void k_max(const float* __restrict__ t2, float* __restrict__ outv) {
    __shared__ float red[256];
    int tid = threadIdx.x;
    float mx = -3.0e38f;
    for (int i = tid; i < N_NODES; i += 256) mx = fmaxf(mx, t2[i]);
    red[tid] = mx;
    __syncthreads();
    for (int s = 128; s > 0; s >>= 1) {
        if (tid < s) red[tid] = fmaxf(red[tid], red[tid + s]);
        __syncthreads();
    }
    if (tid == 0) outv[0] = red[0];
}

// ---------------- kernel 3: fused mask + static-max softmax + att@h (WMMA) + ELU --------
// block = 16 rows, 128 threads (4 waves); wave w covers j in {w*32 + 128*t}.
__global__ void __launch_bounds__(128) k_att(const int* __restrict__ adj,
                                             const float* __restrict__ t1g,
                                             const float* __restrict__ t2g,
                                             const float* __restrict__ t2max_p,
                                             const _Float16* __restrict__ ht,
                                             float* __restrict__ out) {
    __shared__ float s_acc[16 * 64];
    __shared__ float s_l[16];
    const int tid  = threadIdx.x;
    const int lane = tid & 31, wv = tid >> 5;
    const int r = lane & 15, g = lane >> 4;
    const int row0 = blockIdx.x * 16;

    for (int i = tid; i < 1024; i += 128) s_acc[i] = 0.f;
    if (tid < 16) s_l[tid] = 0.f;
    __syncthreads();

    const float t1r = t1g[row0 + r];
    const float m   = lrelu(t1r + t2max_p[0]);   // static per-row softmax max (upper bound)
    const int* adjrow = adj + (size_t)(row0 + r) * N_NODES;

    v8f c[4] = {v8f{}, v8f{}, v8f{}, v8f{}};
    float l = 0.f;

    for (int j = wv * 32; j < N_NODES; j += 128) {
        const int ja = j + 8 * g;
        const int jb = j + 16 + 8 * g;
        // adj: 268 MB streamed exactly once -> non-temporal so it never sweeps L2
        v4i A0 = __builtin_nontemporal_load((const v4i*)(adjrow + ja));
        v4i A1 = __builtin_nontemporal_load((const v4i*)(adjrow + ja + 4));
        v4i A2 = __builtin_nontemporal_load((const v4i*)(adjrow + jb));
        v4i A3 = __builtin_nontemporal_load((const v4i*)(adjrow + jb + 4));
        // t2 / h_t: hot, reused by all 512 blocks -> regular (L1/L2 resident)
        v4f T0 = *(const v4f*)(t2g + ja);
        v4f T1 = *(const v4f*)(t2g + ja + 4);
        v4f T2 = *(const v4f*)(t2g + jb);
        v4f T3 = *(const v4f*)(t2g + jb + 4);

        int   adv[16];
        float tv[16];
#pragma unroll
        for (int i = 0; i < 4; ++i) {
            adv[i] = A0[i];  adv[4 + i] = A1[i];  adv[8 + i] = A2[i];  adv[12 + i] = A3[i];
            tv[i]  = T0[i];  tv[4 + i]  = T1[i];  tv[8 + i]  = T2[i];  tv[12 + i]  = T3[i];
        }

        v16h a;
#pragma unroll
        for (int i = 0; i < 16; ++i) {
            float s = t1r + tv[i];
            s = s >= 0.f ? s : GAT_ALPHA * s;
            s = adv[i] > 0 ? s : GAT_NEG_INF;
            float p = __expf(s - m);       // <= 1, masked -> 0
            l += p;
            a[i] = (_Float16)p;
        }

#pragma unroll
        for (int n = 0; n < 4; ++n) {
            const _Float16* htr = ht + (size_t)(16 * n + r) * N_NODES;
            v8h blo = *(const v8h*)(htr + ja);
            v8h bhi = *(const v8h*)(htr + jb);
            v16h b = __builtin_shufflevector(blo, bhi, 0,1,2,3,4,5,6,7,8,9,10,11,12,13,14,15);
            c[n] = __builtin_amdgcn_wmma_f32_16x16x32_f16(false, a, false, b, (short)0, c[n], false, false);
        }
    }

    // full row denominator for this wave's j-subset (combine K-halves across lane pair)
    l += __shfl_xor(l, 16, 32);
    if (lane < 16) atomicAdd(&s_l[r], l);

    // D[v] at lane = h'[row0 + v + 8g, 16n + r] (partial over this wave's j-subset)
#pragma unroll
    for (int n = 0; n < 4; ++n)
#pragma unroll
        for (int v = 0; v < 8; ++v)
            atomicAdd(&s_acc[(v + 8 * g) * 64 + 16 * n + r], c[n][v]);
    __syncthreads();

    for (int i = tid; i < 1024; i += 128) {
        int row = i >> 6;
        float val = s_acc[i] / s_l[row];
        float o = val > 0.f ? val : (__expf(val) - 1.f);
        // output written once -> non-temporal store
        __builtin_nontemporal_store(o, out + (size_t)(row0 + row) * 64 + (i & 63));
    }
}

// ---------------- host launcher ----------------
extern "C" void kernel_launch(void* const* d_in, const int* in_sizes, int n_in,
                              void* d_out, int out_size, void* d_ws, size_t ws_size,
                              hipStream_t stream) {
    (void)in_sizes; (void)n_in; (void)out_size; (void)ws_size;
    const float* x   = (const float*)d_in[0];
    const float* W   = (const float*)d_in[1];
    const float* a   = (const float*)d_in[2];
    const int*   adj = (const int*)d_in[3];
    float* out = (float*)d_out;

    char* ws = (char*)d_ws;
    float*    t1    = (float*)ws;                               // 8192 f32
    float*    t2    = t1 + N_NODES;                             // 8192 f32
    float*    t2max = t2 + N_NODES;                             // 1 f32 (at 64 KiB)
    _Float16* ht    = (_Float16*)(ws + 65792);                  // 64x8192 f16 = 1 MiB
    _Float16* Wt    = (_Float16*)(ws + 65792 + (size_t)FOUT * N_NODES * 2); // 64x512 f16

    k_wt <<<128, 256, 0, stream>>>(W, Wt);
    k_h  <<<N_NODES / 16, 128, 0, stream>>>(x, Wt, a, t1, t2, ht);
    k_max<<<1, 256, 0, stream>>>(t2, t2max);
    k_att<<<N_NODES / 16, 128, 0, stream>>>(adj, t1, t2, t2max, ht, out);
}
